// LMGroupedQueryAttention_68453188764347
// MI455X (gfx1250) — matrix operand order repaired and checked by reading, not compiled, not run on hardware
//
#include <hip/hip_runtime.h>
#include <hip/hip_bf16.h>

#define T_SEQ  2048
#define DMODEL 2048
#define NH     32
#define NKV    8
#define HD     64
#define DKV    (NKV * HD)   // 512

typedef __attribute__((ext_vector_type(8)))  __bf16 v8bf;
typedef __attribute__((ext_vector_type(16))) __bf16 v16bf;
typedef __attribute__((ext_vector_type(8)))  float  v8f;
typedef __attribute__((ext_vector_type(4)))  unsigned int u32x4;
typedef __attribute__((ext_vector_type(8)))  unsigned int u32x8;

static __device__ __forceinline__ v8f wmma_bf16(v16bf a, v16bf b, v8f c) {
  // D = A(16x32) * B(32x16) + C, f32 accumulate
  return __builtin_amdgcn_wmma_f32_16x16x32_bf16(false, a, false, b, (short)0, c,
                                                 false, false);
}

// A-fragment (16x32 MxK) from row-major memory, stride ld elements.
// ISA layout: lane L holds row M=L%16; K = {8*(L/16)..+7} U {16+8*(L/16)..+7}.
static __device__ __forceinline__ v16bf load_a_frag(const __bf16* tile, int ld) {
  const int lane = threadIdx.x & 31;
  const int row  = lane & 15;
  const int kb   = (lane >> 4) << 3;            // 0 or 8
  const __bf16* p = tile + (size_t)row * ld + kb;
  v8bf lo = *(const v8bf*)(p);
  v8bf hi = *(const v8bf*)(p + 16);
  return __builtin_shufflevector(lo, hi, 0,1,2,3,4,5,6,7,8,9,10,11,12,13,14,15);
}

// B-fragment (32x16 KxN) from N-major memory (row n = contiguous K), stride ld.
// ISA layout: lane L holds col N=L%16; K = {8*(L/16)..+7} U {16+8*(L/16)..+7}.
static __device__ __forceinline__ v16bf load_b_frag(const __bf16* tile, int ld) {
  const int lane = threadIdx.x & 31;
  const __bf16* p = tile + (size_t)(lane & 15) * ld + ((lane >> 4) << 3);
  v8bf lo = *(const v8bf*)(p);
  v8bf hi = *(const v8bf*)(p + 16);
  return __builtin_shufflevector(lo, hi, 0,1,2,3,4,5,6,7,8,9,10,11,12,13,14,15);
}

// ---------------------------------------------------------------------------
// TDM: stage a 2D bf16 tile (tile_d1 rows x tile_d0 elems, row stride in
// elements) from global memory into LDS at lds_byte_addr. D# per ISA 8.3-8.6.
// Issued once per wave; completion tracked by TENSORcnt.
// ---------------------------------------------------------------------------
static __device__ __forceinline__ void tdm_load_2d_bf16(
    unsigned lds_byte_addr, const void* gptr,
    unsigned tile_d0, unsigned tile_d1, unsigned row_stride_elems) {
  unsigned long long ga = (unsigned long long)(size_t)gptr;
  u32x4 g0;
  g0[0] = 1u;                                            // count=1 (valid), user
  g0[1] = lds_byte_addr;                                 // lds_addr
  g0[2] = (unsigned)ga;                                  // global_addr[31:0]
  g0[3] = (unsigned)((ga >> 32) & 0x01FFFFFFu) | (2u << 30); // addr[56:32]|type=2
  u32x8 g1;
  g1[0] = 1u << 16;                                      // data_size = 2 bytes
  g1[1] = (tile_d0 & 0xFFFFu) << 16;                     // tensor_dim0[15:0]
  g1[2] = (tile_d0 >> 16) | ((tile_d1 & 0xFFFFu) << 16); // td0[31:16]|td1[15:0]
  g1[3] = (tile_d1 >> 16) | (tile_d0 << 16);             // td1[31:16]|tile_dim0
  g1[4] = tile_d1 & 0xFFFFu;                             // tile_dim1 | tile_dim2=0
  g1[5] = row_stride_elems;                              // tensor_dim0_stride[31:0]
  g1[6] = 0u;                                            // stride[47:32]|dim1_stride
  g1[7] = 0u;
  u32x4 gz = {0u, 0u, 0u, 0u};                           // groups 2/3 unused (2D)
  asm volatile("tensor_load_to_lds %0, %1, %2, %3"
               :: "s"(g0), "s"(g1), "s"(gz), "s"(gz)
               : "memory");
}

// ---------------------------------------------------------------------------
// C[M,N] = A[M,K] (bf16 row-major) * Bt[N,K]^T (bf16, N-major over K) -> f32
// Block: 256 thr = 8 waves (2 x 4); block tile 128x256; wave tile 64x64.
// A/B K-slices staged to LDS by the Tensor Data Mover, double buffered.
// ---------------------------------------------------------------------------
__global__ __launch_bounds__(256) void gemm_bf16_f32(const __bf16* __restrict__ A,
                                                     const __bf16* __restrict__ Bt,
                                                     float* __restrict__ C,
                                                     int M, int N, int K) {
  __shared__ __align__(128) __bf16 sA[2][128 * 32];   // 16 KB
  __shared__ __align__(128) __bf16 sB[2][256 * 32];   // 32 KB

  const int lane  = threadIdx.x & 31;
  const int w     = threadIdx.x >> 5;
  const int wm    = w >> 2;                 // 0..1
  const int wn    = w & 3;                  // 0..3
  const int mblk  = blockIdx.y * 128;
  const int nblk  = blockIdx.x * 256;

  const __bf16* Ablk = A + (size_t)mblk * K;
  const __bf16* Bblk = Bt + (size_t)nblk * K;
  const unsigned ldsA0 = (unsigned)(size_t)&sA[0][0];
  const unsigned ldsA1 = (unsigned)(size_t)&sA[1][0];
  const unsigned ldsB0 = (unsigned)(size_t)&sB[0][0];
  const unsigned ldsB1 = (unsigned)(size_t)&sB[1][0];

  const v8f zero = {0.f,0.f,0.f,0.f,0.f,0.f,0.f,0.f};
  v8f acc[4][4];
#pragma unroll
  for (int i = 0; i < 4; ++i)
#pragma unroll
    for (int j = 0; j < 4; ++j) acc[i][j] = zero;

  const int nIt = K / 32;
  if (w == 0) {
    tdm_load_2d_bf16(ldsA0, Ablk, 32, 128, (unsigned)K);
    tdm_load_2d_bf16(ldsB0, Bblk, 32, 256, (unsigned)K);
  }

  for (int it = 0; it < nIt; ++it) {
    const int cur = it & 1;
    if (w == 0) __builtin_amdgcn_s_wait_tensorcnt(0);   // current buffer staged
    __syncthreads();
    if (w == 0 && it + 1 < nIt) {                       // prefetch next K slice
      const int k1 = (it + 1) * 32;
      tdm_load_2d_bf16(cur ? ldsA0 : ldsA1, Ablk + k1, 32, 128, (unsigned)K);
      tdm_load_2d_bf16(cur ? ldsB0 : ldsB1, Bblk + k1, 32, 256, (unsigned)K);
    }

    v16bf a[4], b[4];
#pragma unroll
    for (int i = 0; i < 4; ++i)
      a[i] = load_a_frag(&sA[cur][(wm * 64 + 16 * i) * 32], 32);
#pragma unroll
    for (int j = 0; j < 4; ++j)
      b[j] = load_b_frag(&sB[cur][(wn * 64 + 16 * j) * 32], 32);
#pragma unroll
    for (int i = 0; i < 4; ++i)
#pragma unroll
      for (int j = 0; j < 4; ++j) acc[i][j] = wmma_bf16(a[i], b[j], acc[i][j]);

    __syncthreads();   // everyone done with buffer `cur` before it is re-staged
  }

  const int col   = lane & 15;
  const int rbase = (lane >> 4) << 3;       // C layout: VGPR r -> M = r + 8*(lane/16)
  const int mbase = mblk + wm * 64;
  const int nbase = nblk + wn * 64;
#pragma unroll
  for (int i = 0; i < 4; ++i)
#pragma unroll
    for (int j = 0; j < 4; ++j)
#pragma unroll
      for (int r = 0; r < 8; ++r)
        C[(size_t)(mbase + 16 * i + rbase + r) * N + nbase + 16 * j + col] =
            acc[i][j][r];
}

// ---------------------------------------------------------------------------
// fp32 -> bf16 elementwise
// ---------------------------------------------------------------------------
__global__ void cvt_f32_bf16(const float* __restrict__ in, __bf16* __restrict__ out,
                             int n) {
  int i = blockIdx.x * blockDim.x + threadIdx.x;
  if (i < n) out[i] = (__bf16)in[i];
}

// ---------------------------------------------------------------------------
// Tiled transpose + fp32->bf16: W[K,N] f32 -> Wt[N,K] bf16
// ---------------------------------------------------------------------------
__global__ __launch_bounds__(256) void transpose_to_bf16(const float* __restrict__ W,
                                                         __bf16* __restrict__ Wt,
                                                         int K, int N) {
  __shared__ float tile[32][33];
  const int bx = blockIdx.x * 32;           // N direction
  const int by = blockIdx.y * 32;           // K direction
  const int tx = threadIdx.x & 31;
  const int ty = threadIdx.x >> 5;
#pragma unroll
  for (int i = ty; i < 32; i += 8)
    tile[i][tx] = W[(size_t)(by + i) * N + bx + tx];
  __syncthreads();
#pragma unroll
  for (int i = ty; i < 32; i += 8)
    Wt[(size_t)(bx + i) * K + by + tx] = (__bf16)tile[tx][i];
}

// ---------------------------------------------------------------------------
// RoPE + fp32->bf16 (+ optional scale folded in, used for the 1/sqrt(hd) on Q)
// ---------------------------------------------------------------------------
__global__ void rope_cvt_bf16(const float* __restrict__ in,
                              const float* __restrict__ cosp,
                              const float* __restrict__ sinp,
                              __bf16* __restrict__ out, int width, float scale) {
  int idx = blockIdx.x * blockDim.x + threadIdx.x;
  int total = T_SEQ * width;
  if (idx >= total) return;
  int t = idx / width;
  int c = idx - t * width;
  int d = c & 63;
  int hb = c - d;                            // head base (h*64)
  float cv = cosp[t * HD + d];
  float sv = sinp[t * HD + d];
  float v  = in[idx];
  float r;
  if (d < 32) r = v * cv - in[(size_t)t * width + hb + d + 32] * sv;
  else        r = v * cv + in[(size_t)t * width + hb + d - 32] * sv;
  out[idx] = (__bf16)(r * scale);
}

// ---------------------------------------------------------------------------
// Flash attention (causal, GQA). One wave per (16-query tile, head).
// Qb [T, 2048] bf16 (pre-scaled), Kb [T, 512] bf16, Vt [512, T] bf16,
// Yb [T, 2048] bf16 output.
// ---------------------------------------------------------------------------
__global__ __launch_bounds__(256) void attn_flash(const __bf16* __restrict__ Qb,
                                                  const __bf16* __restrict__ Kb,
                                                  const __bf16* __restrict__ Vt,
                                                  const int* __restrict__ amask,
                                                  __bf16* __restrict__ Yb) {
  __shared__ __align__(32) __bf16 sP[8][16 * 32];   // per-wave P staging (1KB each)

  const int lane  = threadIdx.x & 31;
  const int w     = threadIdx.x >> 5;
  const int unit  = blockIdx.x * 8 + w;       // 0 .. 4095
  const int qt    = unit >> 5;                // query tile 0..127
  const int h     = unit & 31;                // head
  const int kh    = h >> 2;                   // kv head (G=4)
  const int q0    = qt * 16;
  const int col   = lane & 15;
  const int rbase = (lane >> 4) << 3;

  const v16bf qa0 = load_a_frag(Qb + (size_t)q0 * DMODEL + h * HD, DMODEL);
  const v16bf qa1 = load_a_frag(Qb + (size_t)q0 * DMODEL + h * HD + 32, DMODEL);

  const v8f zero = {0.f,0.f,0.f,0.f,0.f,0.f,0.f,0.f};
  v8f yacc[4];
#pragma unroll
  for (int j = 0; j < 4; ++j) yacc[j] = zero;
  float mrow[8], lrow[8];
#pragma unroll
  for (int r = 0; r < 8; ++r) { mrow[r] = -1e30f; lrow[r] = 0.f; }

  const int kend = q0 + 16;                   // causal upper bound (exclusive)
  for (int kb = 0; kb < kend; kb += 32) {
    v8f s0 = zero, s1 = zero;
    s0 = wmma_bf16(qa0, load_b_frag(Kb + (size_t)kb * DKV + kh * HD, DKV), s0);
    s0 = wmma_bf16(qa1, load_b_frag(Kb + (size_t)kb * DKV + kh * HD + 32, DKV), s0);
    s1 = wmma_bf16(qa0, load_b_frag(Kb + (size_t)(kb + 16) * DKV + kh * HD, DKV), s1);
    s1 = wmma_bf16(qa1, load_b_frag(Kb + (size_t)(kb + 16) * DKV + kh * HD + 32, DKV), s1);

    const float pad0 = amask[kb + col]      ? 0.f : -1e30f;
    const float pad1 = amask[kb + 16 + col] ? 0.f : -1e30f;
    const bool tail = (kb + 32 > q0);
#pragma unroll
    for (int r = 0; r < 8; ++r) {
      const int qrow = q0 + rbase + r;
      float v0 = s0[r] + pad0;
      float v1 = s1[r] + pad1;
      if (tail) {
        if (kb + col > qrow)      v0 = -1e30f;
        if (kb + 16 + col > qrow) v1 = -1e30f;
      }
      s0[r] = v0; s1[r] = v1;
    }

    float tmax[8];
#pragma unroll
    for (int r = 0; r < 8; ++r) tmax[r] = fmaxf(s0[r], s1[r]);
#pragma unroll
    for (int off = 8; off >= 1; off >>= 1)
#pragma unroll
      for (int r = 0; r < 8; ++r)
        tmax[r] = fmaxf(tmax[r], __shfl_xor(tmax[r], off, 32));

    float tsum[8];
#pragma unroll
    for (int r = 0; r < 8; ++r) {
      const float mn   = fmaxf(mrow[r], tmax[r]);
      const float corr = __expf(mrow[r] - mn);
      mrow[r] = mn;
      const float p0 = __expf(s0[r] - mn);
      const float p1 = __expf(s1[r] - mn);
      s0[r] = p0; s1[r] = p1;
      tsum[r] = p0 + p1;
      lrow[r] *= corr;
#pragma unroll
      for (int j = 0; j < 4; ++j) yacc[j][r] *= corr;
    }
#pragma unroll
    for (int off = 8; off >= 1; off >>= 1)
#pragma unroll
      for (int r = 0; r < 8; ++r) tsum[r] += __shfl_xor(tsum[r], off, 32);
#pragma unroll
    for (int r = 0; r < 8; ++r) lrow[r] += tsum[r];

    // stage P (D-layout) -> LDS -> A-layout; wave-private, DS is in-order
#pragma unroll
    for (int r = 0; r < 8; ++r) {
      sP[w][(rbase + r) * 32 + col]      = (__bf16)s0[r];
      sP[w][(rbase + r) * 32 + 16 + col] = (__bf16)s1[r];
    }
    const v16bf pf = load_a_frag(&sP[w][0], 32);

#pragma unroll
    for (int j = 0; j < 4; ++j) {
      v16bf vf = load_b_frag(Vt + (size_t)(kh * HD + j * 16) * T_SEQ + kb, T_SEQ);
      yacc[j] = wmma_bf16(pf, vf, yacc[j]);
    }
  }

  float inv[8];
#pragma unroll
  for (int r = 0; r < 8; ++r) inv[r] = 1.0f / lrow[r];
#pragma unroll
  for (int j = 0; j < 4; ++j)
#pragma unroll
    for (int r = 0; r < 8; ++r)
      Yb[(size_t)(q0 + rbase + r) * DMODEL + h * HD + j * 16 + col] =
          (__bf16)(yacc[j][r] * inv[r]);
}

// ---------------------------------------------------------------------------
extern "C" void kernel_launch(void* const* d_in, const int* in_sizes, int n_in,
                              void* d_out, int out_size, void* d_ws, size_t ws_size,
                              hipStream_t stream) {
  const float* x    = (const float*)d_in[0];   // [1,T,D]
  const float* cosp = (const float*)d_in[1];   // [1,1,T,64]
  const float* sinp = (const float*)d_in[2];   // [1,1,T,64]
  const int*   am   = (const int*)d_in[3];     // [1,T]
  const float* Wq   = (const float*)d_in[4];   // [D,D]   (in,out)
  const float* Wk   = (const float*)d_in[5];   // [D,512]
  const float* Wv   = (const float*)d_in[6];   // [D,512]
  const float* Wo   = (const float*)d_in[7];   // [D,D]

  char* ws = (char*)d_ws;
  size_t off = 0;
  auto take = [&](size_t bytes) {
    char* p = ws + off;
    off += (bytes + 255) & ~(size_t)255;
    return p;
  };

  __bf16* xb  = (__bf16*)take((size_t)T_SEQ * DMODEL * 2);
  __bf16* wqT = (__bf16*)take((size_t)DMODEL * DMODEL * 2);
  __bf16* wkT = (__bf16*)take((size_t)DMODEL * DKV * 2);
  __bf16* wvT = (__bf16*)take((size_t)DMODEL * DKV * 2);
  __bf16* woT = (__bf16*)take((size_t)DMODEL * DMODEL * 2);
  float*  qf  = (float*)take((size_t)T_SEQ * DMODEL * 4);
  float*  kf  = (float*)take((size_t)T_SEQ * DKV * 4);
  float*  vf  = (float*)take((size_t)T_SEQ * DKV * 4);
  __bf16* qb  = (__bf16*)take((size_t)T_SEQ * DMODEL * 2);
  __bf16* kbf = (__bf16*)take((size_t)T_SEQ * DKV * 2);
  __bf16* vt  = (__bf16*)take((size_t)DKV * T_SEQ * 2);
  __bf16* yb  = (__bf16*)take((size_t)T_SEQ * DMODEL * 2);
  (void)ws_size; (void)in_sizes; (void)n_in; (void)out_size;

  // 1) convert x to bf16
  {
    int n = T_SEQ * DMODEL;
    cvt_f32_bf16<<<(n + 255) / 256, 256, 0, stream>>>(x, xb, n);
  }
  // 2) transpose+convert weights: W[in,out] f32 -> Wt[out,in] bf16
  transpose_to_bf16<<<dim3(DMODEL / 32, DMODEL / 32), 256, 0, stream>>>(Wq, wqT, DMODEL, DMODEL);
  transpose_to_bf16<<<dim3(DKV / 32,    DMODEL / 32), 256, 0, stream>>>(Wk, wkT, DMODEL, DKV);
  transpose_to_bf16<<<dim3(DKV / 32,    DMODEL / 32), 256, 0, stream>>>(Wv, wvT, DMODEL, DKV);
  transpose_to_bf16<<<dim3(DMODEL / 32, DMODEL / 32), 256, 0, stream>>>(Wo, woT, DMODEL, DMODEL);

  // 3) projections (WMMA, TDM-staged)
  gemm_bf16_f32<<<dim3(DMODEL / 256, T_SEQ / 128), 256, 0, stream>>>(xb, wqT, qf, T_SEQ, DMODEL, DMODEL);
  gemm_bf16_f32<<<dim3(DKV / 256,    T_SEQ / 128), 256, 0, stream>>>(xb, wkT, kf, T_SEQ, DKV, DMODEL);
  gemm_bf16_f32<<<dim3(DKV / 256,    T_SEQ / 128), 256, 0, stream>>>(xb, wvT, vf, T_SEQ, DKV, DMODEL);

  // 4) RoPE + bf16 (Q gets 1/sqrt(64) folded in); V transposed for PV WMMA
  {
    int nq = T_SEQ * DMODEL;
    rope_cvt_bf16<<<(nq + 255) / 256, 256, 0, stream>>>(qf, cosp, sinp, qb, DMODEL, 0.125f);
    int nk = T_SEQ * DKV;
    rope_cvt_bf16<<<(nk + 255) / 256, 256, 0, stream>>>(kf, cosp, sinp, kbf, DKV, 1.0f);
  }
  transpose_to_bf16<<<dim3(DKV / 32, T_SEQ / 32), 256, 0, stream>>>(vf, vt, T_SEQ, DKV);

  // 5) flash attention: 128 query tiles x 32 heads = 4096 waves, 8 waves/block
  attn_flash<<<(128 * NH) / 8, 256, 0, stream>>>(qb, kbf, vt, am, yb);

  // 6) output projection -> f32 d_out
  gemm_bf16_f32<<<dim3(DMODEL / 256, T_SEQ / 128), 256, 0, stream>>>(
      yb, woT, (float*)d_out, T_SEQ, DMODEL, DMODEL);
}
